// SetAbstraction_83408264888939
// MI455X (gfx1250) — matrix operand order repaired and checked by reading, not compiled, not run on hardware
//
#include <hip/hip_runtime.h>
#include <hip/hip_bf16.h>

typedef unsigned short u16;
typedef unsigned int   u32;

typedef __attribute__((ext_vector_type(16))) __bf16 v16bf;
typedef __attribute__((ext_vector_type(8)))  float  v8f;

union ABf {
    v16bf v;
    u16   u[16];
    uint4 q[2];
};

__device__ __forceinline__ float bf2f(u16 u) {
    return __uint_as_float(((u32)u) << 16);
}
__device__ __forceinline__ u16 f2bf(float f) {
    u32 u = __float_as_uint(f);
    u += 0x7fffu + ((u >> 16) & 1u);   // round-to-nearest-even
    return (u16)(u >> 16);
}

// ---------------------------------------------------------------------------
// KNN: per-thread register-resident sorted top-32, candidates tiled via LDS.
// Downstream math is permutation-invariant over the neighbor set, so set
// equality with the reference top-k is sufficient.
// ---------------------------------------------------------------------------
#define KNN_K 32

__global__ __launch_bounds__(256) void knn_kernel(const float* __restrict__ xyz,
                                                  int* __restrict__ idx, int N) {
    __shared__ float tsx[256], tsy[256], tsz[256];
    const int tid = threadIdx.x;
    const int q   = blockIdx.x * 256 + tid;     // global query (b*N+n)
    const int b   = q / N;
    const int n   = q - b * N;

    const float qx = xyz[(size_t)q * 3 + 0];
    const float qy = xyz[(size_t)q * 3 + 1];
    const float qz = xyz[(size_t)q * 3 + 2];

    float bd[KNN_K];
    int   bi[KNN_K];
#pragma unroll
    for (int s = 0; s < KNN_K; ++s) { bd[s] = 3.4e38f; bi[s] = 0; }

    for (int t0 = 0; t0 < N; t0 += 256) {
        __syncthreads();
        const int jl = t0 + tid;
        tsx[tid] = xyz[((size_t)b * N + jl) * 3 + 0];
        tsy[tid] = xyz[((size_t)b * N + jl) * 3 + 1];
        tsz[tid] = xyz[((size_t)b * N + jl) * 3 + 2];
        __syncthreads();
        for (int jj = 0; jj < 256; ++jj) {
            const float dx = tsx[jj] - qx;
            const float dy = tsy[jj] - qy;
            const float dz = tsz[jj] - qz;
            const float d  = dx * dx + dy * dy + dz * dz;
            const int   jg = t0 + jj;
            if (jg == n) continue;               // exclude self
            if (d < bd[KNN_K - 1]) {             // branchless shift-insert
#pragma unroll
                for (int s = KNN_K - 1; s >= 1; --s) {
                    const bool shift = d < bd[s - 1];
                    const bool place = (!shift) && (d < bd[s]);
                    const float pd = bd[s - 1];
                    const int   pi = bi[s - 1];
                    bd[s] = shift ? pd : (place ? d : bd[s]);
                    bi[s] = shift ? pi : (place ? jg : bi[s]);
                }
                if (d < bd[0]) { bd[0] = d; bi[0] = jg; }
            }
        }
    }
#pragma unroll
    for (int s = 0; s < KNN_K; ++s) idx[(size_t)q * KNN_K + s] = bi[s];
}

// ---------------------------------------------------------------------------
// Gather: build layer-0 inputs as bf16, channel-padded 6 -> 32 so every GEMM
// uses k-steps of 32 (WMMA bf16 K) and 16B-aligned A loads.
// Rows 0..M0-1: fn rows (b,n,k); rows M0..M0+BN-1: fc rows (b,n).
// ---------------------------------------------------------------------------
__global__ __launch_bounds__(256) void gather_kernel(
        const float* __restrict__ xyz, const float* __restrict__ fea,
        const int* __restrict__ idx, u16* __restrict__ Xfn, u16* __restrict__ Xfc,
        int N, int M0, int BN) {
    const int rid = blockIdx.x * 256 + threadIdx.x;
    if (rid < M0) {
        const int b = rid / (N * 32);
        const int r = rid - b * N * 32;
        const int n = r >> 5;
        const int j = idx[rid];
        const size_t cb = ((size_t)b * N + n) * 3;
        const size_t nb = ((size_t)b * N + j) * 3;
        u16* o = Xfn + (size_t)rid * 32;
        o[0] = f2bf(xyz[nb + 0] - xyz[cb + 0]);
        o[1] = f2bf(xyz[nb + 1] - xyz[cb + 1]);
        o[2] = f2bf(xyz[nb + 2] - xyz[cb + 2]);
        o[3] = f2bf(fea[nb + 0]);
        o[4] = f2bf(fea[nb + 1]);
        o[5] = f2bf(fea[nb + 2]);
#pragma unroll
        for (int c = 6; c < 32; ++c) o[c] = 0;
    } else if (rid < M0 + BN) {
        const int p = rid - M0;
        u16* o = Xfc + (size_t)p * 32;
        o[0] = 0; o[1] = 0; o[2] = 0;
        o[3] = f2bf(fea[(size_t)p * 3 + 0]);
        o[4] = f2bf(fea[(size_t)p * 3 + 1]);
        o[5] = f2bf(fea[(size_t)p * 3 + 2]);
#pragma unroll
        for (int c = 6; c < 32; ++c) o[c] = 0;
    }
}

// ---------------------------------------------------------------------------
// Generic bf16 WMMA GEMM: Y[M,COUT] = relu(bnA*X + bnB) @ W^T (+ bias)
// - bnA==null: raw X (layer 0 / identity)
// - statSum/statSqs != null: accumulate per-output-channel sum & sum-of-squares
//   (register accumulation -> one LDS atomic per N-tile -> one global atomic
//   per channel per block)
// Weights staged once per block into LDS as bf16, zero-padded to CIN cols.
// One wave computes a 16xCOUT tile per iteration (grid-stride over M/16);
// the launch sizes the grid so each wave handles exactly one tile.
// Epilogue uses a single base pointer per N-tile with compile-time offsets so
// the compiler folds store addressing into instruction immediates.
// ---------------------------------------------------------------------------
template <int CIN, int COUT>
__global__ __launch_bounds__(256) void gemm_bn_kernel(
        const u16* __restrict__ X,
        const float* __restrict__ bnA, const float* __restrict__ bnB,
        const float* __restrict__ W, int wCols,
        const float* __restrict__ bias,
        u16* __restrict__ Y,
        float* __restrict__ statSum, float* __restrict__ statSqs,
        int M) {
    __shared__ u16   Wlds[COUT * CIN];
    __shared__ float bnAs[CIN], bnBs[CIN];
    __shared__ float sSum[COUT], sSqs[COUT];

    const int tid = threadIdx.x;
    for (int i = tid; i < COUT * CIN; i += 256) {
        const int n = i / CIN;
        const int k = i - n * CIN;
        const float w = (k < wCols) ? W[n * wCols + k] : 0.0f;
        Wlds[i] = f2bf(w);
    }
    if (bnA) {
        for (int i = tid; i < CIN; i += 256) { bnAs[i] = bnA[i]; bnBs[i] = bnB[i]; }
    }
    if (statSum) {
        for (int i = tid; i < COUT; i += 256) { sSum[i] = 0.0f; sSqs[i] = 0.0f; }
    }
    __syncthreads();

    const int wave = tid >> 5;
    const int lane = tid & 31;
    const int l15  = lane & 15;
    const int half = lane >> 4;
    constexpr int NT = COUT / 16;
    constexpr int KS = CIN / 32;
    const int tiles = M >> 4;

    // channel per N-tile is tile-invariant: hoist bias, accumulate stats in regs
    float bsv[NT], accS[NT], accQ[NT];
#pragma unroll
    for (int t = 0; t < NT; ++t) {
        bsv[t]  = bias ? bias[16 * t + l15] : 0.0f;
        accS[t] = 0.0f;
        accQ[t] = 0.0f;
    }

    for (int mt = blockIdx.x * 8 + wave; mt < tiles; mt += gridDim.x * 8) {
        const int mBase = mt << 4;
        v8f acc[NT];
#pragma unroll
        for (int t = 0; t < NT; ++t)
#pragma unroll
            for (int v = 0; v < 8; ++v) acc[t][v] = 0.0f;

#pragma unroll
        for (int ks = 0; ks < KS; ++ks) {
            // A fragment: 16-bit A 16x32 layout (ISA 7.12.2):
            // lane l15 = row, element e -> K = ks*32 + ((e&8)<<1) + half*8 + (e&7)
            ABf af;
            const u16* xr = X + (size_t)(mBase + l15) * CIN + ks * 32 + half * 8;
            af.q[0] = *(const uint4*)xr;
            af.q[1] = *(const uint4*)(xr + 16);
            if (bnA) {
#pragma unroll
                for (int e = 0; e < 16; ++e) {
                    const int k = ks * 32 + ((e & 8) << 1) + half * 8 + (e & 7);
                    float f = bf2f(af.u[e]);
                    f = fmaxf(bnAs[k] * f + bnBs[k], 0.0f);
                    af.u[e] = f2bf(f);
                }
            }
            // B fragments: lane = column n, K same as A: ((e&8)<<1) + half*8 + (e&7), from LDS
#pragma unroll
            for (int t = 0; t < NT; ++t) {
                ABf bf_;
                const u16* wr = &Wlds[(l15 + 16 * t) * CIN + ks * 32 + half * 8];
                bf_.q[0] = *(const uint4*)wr;
                bf_.q[1] = *(const uint4*)(wr + 16);
                acc[t] = __builtin_amdgcn_wmma_f32_16x16x32_bf16(
                    false, af.v, false, bf_.v, (short)0, acc[t], false, false);
            }
        }

        // D layout: element v -> M = mBase + half*8 + v, N = 16*t + l15
        // One base pointer per N-tile; v*COUT folds into instruction offsets.
        u16* yrow = Y + (size_t)(mBase + half * 8) * COUT + l15;
#pragma unroll
        for (int t = 0; t < NT; ++t) {
            u16* yp = yrow + 16 * t;
            const float bs = bsv[t];
            float s = 0.0f, ss = 0.0f;
#pragma unroll
            for (int v = 0; v < 8; ++v) {
                const float y = acc[t][v] + bs;
                yp[v * COUT] = f2bf(y);
                s += y;
                ss += y * y;
            }
            accS[t] += s;
            accQ[t] += ss;
        }
    }

    if (statSum) {
#pragma unroll
        for (int t = 0; t < NT; ++t) {
            atomicAdd(&sSum[16 * t + l15], accS[t]);
            atomicAdd(&sSqs[16 * t + l15], accQ[t]);
        }
    }
    __syncthreads();
    if (statSum) {
        for (int i = tid; i < COUT; i += 256) {
            atomicAdd(&statSum[i], sSum[i]);
            atomicAdd(&statSqs[i], sSqs[i]);
        }
    }
}

// ---------------------------------------------------------------------------
// Fold BN stats into per-channel scale/shift:
//   relu(gamma*(y-mean)*rsqrt(var+eps)+beta) == relu(a*y + b)
// ---------------------------------------------------------------------------
__global__ void bn_finalize_kernel(const float* __restrict__ sum,
                                   const float* __restrict__ sqs,
                                   const float* __restrict__ gamma,
                                   const float* __restrict__ beta,
                                   float invM, float* __restrict__ outA,
                                   float* __restrict__ outB, int C) {
    const int c = blockIdx.x * blockDim.x + threadIdx.x;
    if (c < C) {
        const float mean = sum[c] * invM;
        const float var  = sqs[c] * invM - mean * mean;  // biased, matches jnp.var
        const float r    = rsqrtf(var + 1e-5f);
        const float a    = gamma[c] * r;
        outA[c] = a;
        outB[c] = beta[c] - mean * a;
    }
}

// ---------------------------------------------------------------------------
// Attention combine: one wave per point. lane j computes the neighbor-j logit
// dot(qc, kn_j)/sqrt(64); wave-wide softmax over 32 neighbors + center; then
// lanes switch to channel-parallel weighted sum over V (coalesced bf16 reads).
// ---------------------------------------------------------------------------
__global__ __launch_bounds__(256) void attn_combine_kernel(
        const u16* __restrict__ qc, const u16* __restrict__ kc,
        const u16* __restrict__ vc, const u16* __restrict__ kn,
        const u16* __restrict__ vn, float* __restrict__ out) {
    __shared__ float qcs[8][64];
    __shared__ float wls[8][36];

    const int tid  = threadIdx.x;
    const int wave = tid >> 5;
    const int lane = tid & 31;
    const size_t p = (size_t)blockIdx.x * 8 + wave;

    // stage qc (f32) into LDS; center logit from qc.kc
    const u32 qu = ((const u32*)(qc + p * 64))[lane];
    const float q0 = bf2f((u16)qu), q1 = bf2f((u16)(qu >> 16));
    qcs[wave][2 * lane]     = q0;
    qcs[wave][2 * lane + 1] = q1;
    const u32 ku = ((const u32*)(kc + p * 64))[lane];
    float part = q0 * bf2f((u16)ku) + q1 * bf2f((u16)(ku >> 16));
#pragma unroll
    for (int o = 16; o; o >>= 1) part += __shfl_xor(part, o, 32);
    const float wc = part * 0.125f;   // 1/sqrt(64)
    __syncthreads();

    // neighbor logits
    const u16* kr = kn + (p * 32 + lane) * 64;
    float dot = 0.0f;
#pragma unroll
    for (int c8 = 0; c8 < 8; ++c8) {
        const uint4 kq = ((const uint4*)kr)[c8];
        const u32 wv4[4] = {kq.x, kq.y, kq.z, kq.w};
#pragma unroll
        for (int h = 0; h < 4; ++h) {
            dot += qcs[wave][c8 * 8 + 2 * h]     * bf2f((u16)wv4[h]);
            dot += qcs[wave][c8 * 8 + 2 * h + 1] * bf2f((u16)(wv4[h] >> 16));
        }
    }
    const float wj = dot * 0.125f;

    // softmax over 33 (32 lanes + center)
    float mx = wj;
#pragma unroll
    for (int o = 16; o; o >>= 1) mx = fmaxf(mx, __shfl_xor(mx, o, 32));
    mx = fmaxf(mx, wc);
    const float ej = __expf(wj - mx);
    float ssum = ej;
#pragma unroll
    for (int o = 16; o; o >>= 1) ssum += __shfl_xor(ssum, o, 32);
    const float ec = __expf(wc - mx);
    const float inv = 1.0f / (ssum + ec);
    wls[wave][lane] = ej * inv;
    if (lane == 0) wls[wave][32] = ec * inv;
    __syncthreads();

    // weighted sum of V (lane -> channels lane, lane+32, lane+64, lane+96)
    const float wcn = wls[wave][32];
    float a0 = wcn * bf2f(vc[p * 128 + lane]);
    float a1 = wcn * bf2f(vc[p * 128 + 32 + lane]);
    float a2 = wcn * bf2f(vc[p * 128 + 64 + lane]);
    float a3 = wcn * bf2f(vc[p * 128 + 96 + lane]);
    const u16* vr = vn + p * 32 * 128;
    for (int j = 0; j < 32; ++j) {
        const float w = wls[wave][j];
        const u16* r = vr + j * 128;
        a0 += w * bf2f(r[lane]);
        a1 += w * bf2f(r[lane + 32]);
        a2 += w * bf2f(r[lane + 64]);
        a3 += w * bf2f(r[lane + 96]);
    }
    out[p * 128 + lane]      = a0;
    out[p * 128 + 32 + lane] = a1;
    out[p * 128 + 64 + lane] = a2;
    out[p * 128 + 96 + lane] = a3;
}

// ---------------------------------------------------------------------------
extern "C" void kernel_launch(void* const* d_in, const int* in_sizes, int n_in,
                              void* d_out, int out_size, void* d_ws, size_t ws_size,
                              hipStream_t stream) {
    (void)n_in; (void)out_size; (void)ws_size;
    const float* xyz = (const float*)d_in[0];
    const float* fea = (const float*)d_in[1];
    const float* w0  = (const float*)d_in[2];
    const float* b0  = (const float*)d_in[3];
    const float* g0  = (const float*)d_in[4];
    const float* be0 = (const float*)d_in[5];
    const float* w1  = (const float*)d_in[6];
    const float* b1  = (const float*)d_in[7];
    const float* g1  = (const float*)d_in[8];
    const float* be1 = (const float*)d_in[9];
    const float* w2  = (const float*)d_in[10];
    const float* b2  = (const float*)d_in[11];
    const float* g2  = (const float*)d_in[12];
    const float* be2 = (const float*)d_in[13];
    const float* wq  = (const float*)d_in[14];
    const float* wk  = (const float*)d_in[15];
    const float* wv  = (const float*)d_in[16];

    const int BN = in_sizes[0] / 3;  // B*N (= 16384)
    const int N  = BN / 2;           // B = 2
    const int M0 = BN * KNN_K;       // B*N*K rows

    char* base = (char*)d_ws;
    size_t off = 0;
    auto alloc = [&](size_t bytes) -> void* {
        void* p = base + off;
        off += (bytes + 255) & ~(size_t)255;
        return p;
    };
    int* idx   = (int*)alloc((size_t)M0 * 4);
    u16* X0fn  = (u16*)alloc((size_t)M0 * 32 * 2);
    u16* X0fc  = (u16*)alloc((size_t)BN * 32 * 2);
    u16* y0fn  = (u16*)alloc((size_t)M0 * 64 * 2);
    u16* y0fc  = (u16*)alloc((size_t)BN * 64 * 2);
    u16* y1fn  = (u16*)alloc((size_t)M0 * 64 * 2);
    u16* y1fc  = (u16*)alloc((size_t)BN * 64 * 2);
    u16* y2fn  = (u16*)alloc((size_t)M0 * 128 * 2);
    u16* y2fc  = (u16*)alloc((size_t)BN * 128 * 2);
    u16* vnb   = (u16*)alloc((size_t)M0 * 128 * 2);
    u16* qcb   = (u16*)alloc((size_t)BN * 64 * 2);
    u16* kcb   = (u16*)alloc((size_t)BN * 64 * 2);
    u16* vcb   = (u16*)alloc((size_t)BN * 128 * 2);
    float* st  = (float*)alloc(2048 * 4);
    u16* knb   = y0fn;  // reuse: y0fn is dead before kn is written

    float *sum0fn = st,       *sqs0fn = st + 64,  *sum0fc = st + 128, *sqs0fc = st + 192;
    float *sum1fn = st + 256, *sqs1fn = st + 320, *sum1fc = st + 384, *sqs1fc = st + 448;
    float *sum2fn = st + 512, *sqs2fn = st + 640, *sum2fc = st + 768, *sqs2fc = st + 896;
    float* pp = st + 1024;
    float *a0fn = pp,       *c0fn = pp + 64,  *a0fc = pp + 128, *c0fc = pp + 192;
    float *a1fn = pp + 256, *c1fn = pp + 320, *a1fc = pp + 384, *c1fc = pp + 448;
    float *a2fn = pp + 512, *c2fn = pp + 640, *a2fc = pp + 768, *c2fc = pp + 896;

    hipMemsetAsync(st, 0, 1024 * 4, stream);

    knn_kernel<<<BN / 256, 256, 0, stream>>>(xyz, idx, N);
    gather_kernel<<<(M0 + BN + 255) / 256, 256, 0, stream>>>(xyz, fea, idx, X0fn,
                                                             X0fc, N, M0, BN);

    auto gb = [](int M) {
        int blk = ((M >> 4) + 7) >> 3;
        if (blk < 1) blk = 1;
        if (blk > 4096) blk = 4096;
        return blk;
    };

    // layer 0 (raw bf16 input, real 6 weight cols padded to 32)
    gemm_bn_kernel<32, 64><<<gb(M0), 256, 0, stream>>>(X0fn, nullptr, nullptr, w0, 6,
                                                       b0, y0fn, sum0fn, sqs0fn, M0);
    gemm_bn_kernel<32, 64><<<gb(BN), 256, 0, stream>>>(X0fc, nullptr, nullptr, w0, 6,
                                                       b0, y0fc, sum0fc, sqs0fc, BN);
    bn_finalize_kernel<<<1, 64, 0, stream>>>(sum0fn, sqs0fn, g0, be0, 1.0f / M0, a0fn, c0fn, 64);
    bn_finalize_kernel<<<1, 64, 0, stream>>>(sum0fc, sqs0fc, g0, be0, 1.0f / BN, a0fc, c0fc, 64);
    // layer 1
    gemm_bn_kernel<64, 64><<<gb(M0), 256, 0, stream>>>(y0fn, a0fn, c0fn, w1, 64, b1,
                                                       y1fn, sum1fn, sqs1fn, M0);
    gemm_bn_kernel<64, 64><<<gb(BN), 256, 0, stream>>>(y0fc, a0fc, c0fc, w1, 64, b1,
                                                       y1fc, sum1fc, sqs1fc, BN);
    bn_finalize_kernel<<<1, 64, 0, stream>>>(sum1fn, sqs1fn, g1, be1, 1.0f / M0, a1fn, c1fn, 64);
    bn_finalize_kernel<<<1, 64, 0, stream>>>(sum1fc, sqs1fc, g1, be1, 1.0f / BN, a1fc, c1fc, 64);
    // layer 2
    gemm_bn_kernel<64, 128><<<gb(M0), 256, 0, stream>>>(y1fn, a1fn, c1fn, w2, 64, b2,
                                                        y2fn, sum2fn, sqs2fn, M0);
    gemm_bn_kernel<64, 128><<<gb(BN), 256, 0, stream>>>(y1fc, a1fc, c1fc, w2, 64, b2,
                                                        y2fc, sum2fc, sqs2fc, BN);
    bn_finalize_kernel<<<1, 128, 0, stream>>>(sum2fn, sqs2fn, g2, be2, 1.0f / M0, a2fn, c2fn, 128);
    bn_finalize_kernel<<<1, 128, 0, stream>>>(sum2fc, sqs2fc, g2, be2, 1.0f / BN, a2fc, c2fc, 128);
    // attention projections (BN+ReLU fused into A-load; no bias/stats)
    gemm_bn_kernel<128, 64><<<gb(BN), 256, 0, stream>>>(y2fc, a2fc, c2fc, wq, 128,
                                                        nullptr, qcb, nullptr, nullptr, BN);
    gemm_bn_kernel<128, 64><<<gb(BN), 256, 0, stream>>>(y2fc, a2fc, c2fc, wk, 128,
                                                        nullptr, kcb, nullptr, nullptr, BN);
    gemm_bn_kernel<128, 128><<<gb(BN), 256, 0, stream>>>(y2fc, a2fc, c2fc, wv, 128,
                                                         nullptr, vcb, nullptr, nullptr, BN);
    gemm_bn_kernel<128, 64><<<gb(M0), 256, 0, stream>>>(y2fn, a2fn, c2fn, wk, 128,
                                                        nullptr, knb, nullptr, nullptr, M0);
    gemm_bn_kernel<128, 128><<<gb(M0), 256, 0, stream>>>(y2fn, a2fn, c2fn, wv, 128,
                                                         nullptr, vnb, nullptr, nullptr, M0);
    // softmax + weighted sum -> f32 output (B,N,128)
    attn_combine_kernel<<<BN / 8, 256, 0, stream>>>(qcb, kcb, vcb, knb, vnb,
                                                    (float*)d_out);
}
